// TaskAlignedAssigner_51376398795607
// MI455X (gfx1250) — matrix operand order repaired and checked by reading, not compiled, not run on hardware
//
#include <hip/hip_runtime.h>
#include <hip/hip_bf16.h>

#define EPS   1e-9f
#define TOPK  13

typedef _Float16 half16 __attribute__((ext_vector_type(16)));
typedef float    float8 __attribute__((ext_vector_type(8)));

// GCC-style vector + address-space-qualified pointee types matching the async builtins'
// prototypes (param types confirmed from compiler diagnostics: AS1 int / AS1 int4).
typedef int v4i_t __attribute__((vector_size(16)));
typedef __attribute__((address_space(1))) int   gdev_i32;
typedef __attribute__((address_space(3))) int   glds_i32;
typedef __attribute__((address_space(1))) v4i_t gdev_v4i;
typedef __attribute__((address_space(3))) v4i_t glds_v4i;

#if __has_builtin(__builtin_amdgcn_global_store_async_from_lds_b128) && \
    __has_builtin(__builtin_amdgcn_s_wait_asynccnt)
#define ASYNC_ST 1
#else
#define ASYNC_ST 0
#endif

#if __has_builtin(__builtin_amdgcn_global_load_async_to_lds_b32) && \
    __has_builtin(__builtin_amdgcn_s_wait_asynccnt)
#define ASYNC_LD 1
#else
#define ASYNC_LD 0
#endif

// ---------------- K0: zero-init scratch that is atomically accumulated ----------------
__global__ void k0_init(unsigned int* __restrict__ intopk, int* __restrict__ maxmet,
                        int* __restrict__ maxiou, int nBA, int nBM) {
  int i = blockIdx.x * blockDim.x + threadIdx.x;
  int stride = gridDim.x * blockDim.x;
  for (int j = i; j < nBA; j += stride) intopk[j] = 0u;
  for (int j = i; j < nBM; j += stride) { maxmet[j] = 0; maxiou[j] = 0; }
}

// ---------------- K1: WMMA label-gather  G[b][m][a] = pred_scores[b,a,label[b,m]] ----
// 128 threads = 4 waves; each wave owns a 16-anchor tile, covers all 32 gt columns.
__global__ void k1_gather_scores(const float* __restrict__ pred_scores,
                                 const int* __restrict__ gt_labels,
                                 float* __restrict__ metrics,
                                 int A, int C, int M) {
  __shared__ float tile[32 * 64];  // 32 gts x 64 anchors, 8KB
  const int b     = blockIdx.y;
  const int t     = threadIdx.x;
  const int wave  = t >> 5;
  const int lane  = t & 31;
  const int a0blk = blockIdx.x * 64;
  const int a0    = a0blk + wave * 16;

  if (a0 + 15 < A) {                 // wave-uniform: EXEC stays all-ones for WMMA
    const int row = lane & 15;       // A-matrix row (anchor within tile)
    const int grp = lane >> 4;       // K sub-group select per ISA A/B layouts
    const int a   = a0 + row;
    const float* srow = pred_scores + ((size_t)b * A + a) * C;
    const bool c4 = (C & 3) == 0;

    float8 acc[2];
#pragma unroll
    for (int j = 0; j < 8; ++j) { acc[0][j] = 0.f; acc[1][j] = 0.f; }

    const int col  = lane & 15;
    const int lab0 = (col      < M) ? gt_labels[b * M + col]      : -1;
    const int lab1 = (col + 16 < M) ? gt_labels[b * M + col + 16] : -1;

    for (int kc = 0; kc < C; kc += 32) {
      // A fragment: 16 halves = channels {kc+grp*8..+7, kc+16+grp*8..+7}
      half16 afrag;
      const int cb0 = kc + grp * 8;
      const int cb1 = kc + 16 + grp * 8;
#pragma unroll
      for (int h = 0; h < 2; ++h) {
        const int cbase = h ? cb1 : cb0;
#pragma unroll
        for (int q = 0; q < 2; ++q) {
          const int c0 = cbase + q * 4;
          float4 v;
          if (c4 && (c0 + 3 < C)) {
            v = *(const float4*)(srow + c0);
          } else {
            v.x = (c0 + 0 < C) ? srow[c0 + 0] : 0.f;
            v.y = (c0 + 1 < C) ? srow[c0 + 1] : 0.f;
            v.z = (c0 + 2 < C) ? srow[c0 + 2] : 0.f;
            v.w = (c0 + 3 < C) ? srow[c0 + 3] : 0.f;
          }
          const int o = h * 8 + q * 4;
          afrag[o + 0] = (_Float16)v.x;
          afrag[o + 1] = (_Float16)v.y;
          afrag[o + 2] = (_Float16)v.z;
          afrag[o + 3] = (_Float16)v.w;
        }
      }
      // One-hot B fragments, generated in registers (exact in f16).
      const int kbase = kc + grp * 8;
      half16 bf0, bf1;
#pragma unroll
      for (int i = 0; i < 16; ++i) {
        const int ch = kbase + (i < 8 ? i : i + 8);
        bf0[i] = (_Float16)((lab0 == ch) ? 1.0f : 0.0f);
        bf1[i] = (_Float16)((lab1 == ch) ? 1.0f : 0.0f);
      }
      acc[0] = __builtin_amdgcn_wmma_f32_16x16x32_f16(false, afrag, false, bf0,
                                                      (short)0, acc[0], false, false);
      acc[1] = __builtin_amdgcn_wmma_f32_16x16x32_f16(false, afrag, false, bf1,
                                                      (short)0, acc[1], false, false);
    }
    // D layout: vgpr j, lane l -> anchor = j + 8*(l>>4), gt col = l&15
#pragma unroll
    for (int nt = 0; nt < 2; ++nt)
#pragma unroll
      for (int j = 0; j < 8; ++j) {
        const int m  = nt * 16 + (lane & 15);
        const int al = wave * 16 + j + 8 * grp;
        tile[m * 64 + al] = acc[nt][j];
      }
  }
  __syncthreads();
#if ASYNC_ST
  // Async LDS->global: 32 rows x 16 chunks of 16B, issued per-lane, ASYNCcnt tracked.
  for (int idx = t; idx < 32 * 16; idx += 128) {
    const int m  = idx >> 4;
    const int al = (idx & 15) << 2;
    const int a  = a0blk + al;
    if (m < M) {
      float* gp = metrics + ((size_t)(b * M + m)) * A + a;
      if (a + 3 < A) {
        __builtin_amdgcn_global_store_async_from_lds_b128(
            (gdev_v4i*)gp, (glds_v4i*)&tile[m * 64 + al], 0, 0);
      } else {
#pragma unroll
        for (int q = 0; q < 4; ++q)
          if (a + q < A) gp[q] = tile[m * 64 + al + q];
      }
    }
  }
  __builtin_amdgcn_s_wait_asynccnt(0);
#else
  // Block-wide coalesced store of the 32(m) x 64(a) tile.
  for (int idx = t; idx < 32 * 64; idx += 128) {
    const int m  = idx >> 6;
    const int al = idx & 63;
    const int a  = a0blk + al;
    if (a < A && m < M)
      metrics[((size_t)(b * M + m)) * A + a] = tile[idx];
  }
#endif
}

// Stage 2*M gt-bbox floats into LDS (async global->LDS when available).
__device__ __forceinline__ void stage_gts(const float* __restrict__ gt_bboxes,
                                          float* g, int b, int t, int M) {
#if ASYNC_LD
  if (t < 2 * M)
    __builtin_amdgcn_global_load_async_to_lds_b32(
        (gdev_i32*)(gt_bboxes + b * 2 * M + t), (glds_i32*)&g[t], 0, 0);
  __builtin_amdgcn_s_wait_asynccnt(0);
  __syncthreads();
#else
  if (t < 2 * M) g[t] = gt_bboxes[b * 2 * M + t];
  __syncthreads();
#endif
}

// ---------------- K2: per-anchor IoU, metric = score*iou^6, in-gts bits, argmax-iou --
__global__ void k2_metrics(const float* __restrict__ pred_bboxes,
                           const float* __restrict__ gt_bboxes,
                           float* __restrict__ metrics,
                           unsigned int* __restrict__ ingts,
                           int* __restrict__ bestm,
                           int A, int M) {
  __shared__ float g[64];
  const int b = blockIdx.y;
  const int t = threadIdx.x;
  stage_gts(gt_bboxes, g, b, t, M);
  const int a = blockIdx.x * blockDim.x + t;
  if (a >= A) return;
  const float ps = pred_bboxes[((size_t)b * A + a) * 2 + 0];
  const float pe = pred_bboxes[((size_t)b * A + a) * 2 + 1];
  const float pt = (float)a + 0.5f;
  unsigned int bits = 0u;
  float best = -1.f; int bm = 0;
  for (int m = 0; m < M; ++m) {
    const float gs = g[2 * m], ge = g[2 * m + 1];
    const float inter = fmaxf(fminf(ge, pe) - fmaxf(gs, ps), 0.f);
    const float uni   = (ge - gs) + (pe - ps) - inter;
    const float iou   = inter / (uni + EPS);
    const size_t off  = ((size_t)(b * M + m)) * A + a;
    const float sc    = metrics[off];
    const float i2    = iou * iou;
    metrics[off] = sc * i2 * i2 * i2;
    if ((pt - gs > EPS) && (ge - pt > EPS)) bits |= (1u << m);
    if (iou > best) { best = iou; bm = m; }
  }
  ingts[(size_t)b * A + a] = bits;
  bestm[(size_t)b * A + a] = bm;
}

// ---------------- K3: top-13 per (b,m) row over A anchors -> intopk bitmask ---------
__global__ void k3_topk(const float* __restrict__ metrics,
                        const unsigned int* __restrict__ ingts,
                        unsigned int* __restrict__ intopk,
                        int A, int M) {
  __shared__ float sv[256 * TOPK];
  __shared__ int   si[256 * TOPK];
  __shared__ float rv[256];
  __shared__ int   ri[256];
  const int row = blockIdx.x;          // b*M + m
  const int b = row / M, m = row % M;
  const int t = threadIdx.x;
  const float* met = metrics + (size_t)row * A;
  const unsigned int* ig = ingts + (size_t)b * A;

  float bv[TOPK]; int bi[TOPK];
#pragma unroll
  for (int k = 0; k < TOPK; ++k) { bv[k] = -1.f; bi[k] = -1; }
  for (int a = t; a < A; a += 256) {
    const float v = ((ig[a] >> m) & 1u) ? met[a] : 0.f;
    if (v > bv[TOPK - 1]) {
      int k = TOPK - 1;
      while (k > 0 && bv[k - 1] < v) { bv[k] = bv[k - 1]; bi[k] = bi[k - 1]; --k; }
      bv[k] = v; bi[k] = a;
    }
  }
#pragma unroll
  for (int k = 0; k < TOPK; ++k) { sv[t * TOPK + k] = bv[k]; si[t * TOPK + k] = bi[k]; }
  __syncthreads();

  for (int k = 0; k < TOPK; ++k) {
    float mv = -2.f; int mp = t * TOPK;
#pragma unroll
    for (int j = 0; j < TOPK; ++j) {
      const float v = sv[t * TOPK + j];
      if (v > mv) { mv = v; mp = t * TOPK + j; }
    }
    rv[t] = mv; ri[t] = mp;
    __syncthreads();
    for (int s = 128; s > 0; s >>= 1) {
      if (t < s && rv[t + s] > rv[t]) { rv[t] = rv[t + s]; ri[t] = ri[t + s]; }
      __syncthreads();
    }
    if (t == 0) {
      const int pos  = ri[0];
      const int aidx = si[pos];
      if (aidx >= 0) atomicOr(&intopk[(size_t)b * A + aidx], 1u << m);
      sv[pos] = -2.f;
    }
    __syncthreads();
  }
}

// ---------------- K4: mask_positive, labels/bboxes, per-(b,m) max metric/iou --------
__global__ void k4_assign(const float* __restrict__ pred_bboxes,
                          const float* __restrict__ gt_bboxes,
                          const int* __restrict__ gt_labels,
                          const int* __restrict__ bg_index_p,
                          const float* __restrict__ metrics,
                          const unsigned int* __restrict__ ingts,
                          const unsigned int* __restrict__ intopk,
                          const int* __restrict__ bestm,
                          unsigned int* __restrict__ maskpos,
                          int* __restrict__ maxmet_i,
                          int* __restrict__ maxiou_i,
                          float* __restrict__ out_labels,
                          float* __restrict__ out_bboxes,
                          int A, int M) {
  __shared__ float g[64];
  const int b = blockIdx.y;
  const int t = threadIdx.x;
  stage_gts(gt_bboxes, g, b, t, M);
  const int a = blockIdx.x * blockDim.x + t;
  if (a >= A) return;
  unsigned int mp = ingts[(size_t)b * A + a] & intopk[(size_t)b * A + a];
  if (__popc(mp) > 1) mp = 1u << bestm[(size_t)b * A + a];
  maskpos[(size_t)b * A + a] = mp;
  const int s  = __popc(mp);
  const int ag = s ? (__ffs(mp) - 1) : 0;
  const int lab = s ? gt_labels[b * M + ag] : bg_index_p[0];
  out_labels[(size_t)b * A + a] = (float)lab;
  out_bboxes[((size_t)b * A + a) * 2 + 0] = g[2 * ag];
  out_bboxes[((size_t)b * A + a) * 2 + 1] = g[2 * ag + 1];
  if (mp) {
    const float ps = pred_bboxes[((size_t)b * A + a) * 2 + 0];
    const float pe = pred_bboxes[((size_t)b * A + a) * 2 + 1];
    unsigned int bits = mp;
    while (bits) {
      const int m = __ffs(bits) - 1;
      bits &= bits - 1;
      const float gs = g[2 * m], ge = g[2 * m + 1];
      const float inter = fmaxf(fminf(ge, pe) - fmaxf(gs, ps), 0.f);
      const float uni   = (ge - gs) + (pe - ps) - inter;
      const float iou   = inter / (uni + EPS);
      const float met   = metrics[((size_t)(b * M + m)) * A + a];
      atomicMax(&maxmet_i[b * M + m], __float_as_int(met));  // nonneg floats: monotone
      atomicMax(&maxiou_i[b * M + m], __float_as_int(iou));
    }
  }
}

// ---------------- K5: normalized alignment + coalesced one-hot score rows -----------
__global__ void k5_scores(const unsigned int* __restrict__ maskpos,
                          const float* __restrict__ metrics,
                          const int* __restrict__ maxmet_i,
                          const int* __restrict__ maxiou_i,
                          const float* __restrict__ out_labels,
                          float* __restrict__ out_scores,
                          int A, int C, int M) {
  __shared__ float amv[256];
  __shared__ int   labv[256];
  const int b  = blockIdx.y;
  const int t  = threadIdx.x;
  const int a0 = blockIdx.x * 256;
  const int a  = a0 + t;
  float am = 0.f; int lab = C;
  if (a < A) {
    unsigned int bits = maskpos[(size_t)b * A + a];
    while (bits) {
      const int m = __ffs(bits) - 1;
      bits &= bits - 1;
      const float met = metrics[((size_t)(b * M + m)) * A + a];
      const float mm  = __int_as_float(maxmet_i[b * M + m]);
      const float mi  = __int_as_float(maxiou_i[b * M + m]);
      am = fmaxf(am, met / (mm + EPS) * mi);
    }
    lab = (int)out_labels[(size_t)b * A + a];
  }
  amv[t] = am; labv[t] = lab;
  __syncthreads();
  float* srow = out_scores + ((size_t)b * A + a0) * C;
  const int total = 256 * C;
  for (int idx = t; idx < total; idx += 256) {
    const int al = idx / C;
    const int c  = idx - al * C;
    if (a0 + al < A)
      srow[(size_t)al * C + c] = (c == labv[al]) ? amv[al] : 0.f;
  }
}

extern "C" void kernel_launch(void* const* d_in, const int* in_sizes, int n_in,
                              void* d_out, int out_size, void* d_ws, size_t ws_size,
                              hipStream_t stream) {
  const float* pred_scores = (const float*)d_in[0];
  const float* pred_bboxes = (const float*)d_in[1];
  const int A = in_sizes[2];                 // anchor_points is (A,1)
  const int B = in_sizes[1] / (2 * A);       // pred_bboxes is (B,A,2)
  const int C = in_sizes[0] / (B * A);       // pred_scores is (B,A,C)
  const int M = in_sizes[4] / B;             // gt_labels is (B,M,1)
  const int* gt_labels = (const int*)d_in[4];
  const float* gt_bboxes = (const float*)d_in[5];
  const int* bg_index = (const int*)d_in[7];

  float* out        = (float*)d_out;
  float* out_labels = out;
  float* out_bboxes = out + (size_t)B * A;
  float* out_scores = out + (size_t)B * A * 3;

  char* ws = (char*)d_ws;
  float*        metrics = (float*)ws;        ws += (size_t)B * M * A * 4;
  unsigned int* ingts   = (unsigned int*)ws; ws += (size_t)B * A * 4;
  int*          bestm   = (int*)ws;          ws += (size_t)B * A * 4;
  unsigned int* intopk  = (unsigned int*)ws; ws += (size_t)B * A * 4;
  unsigned int* maskpos = (unsigned int*)ws; ws += (size_t)B * A * 4;
  int*          maxmet  = (int*)ws;          ws += (size_t)B * M * 4;
  int*          maxiou  = (int*)ws;          ws += (size_t)B * M * 4;

  k0_init<<<128, 256, 0, stream>>>(intopk, maxmet, maxiou, B * A, B * M);

  dim3 g1((A + 63) / 64, B);
  k1_gather_scores<<<g1, 128, 0, stream>>>(pred_scores, gt_labels, metrics, A, C, M);

  dim3 g2((A + 255) / 256, B);
  k2_metrics<<<g2, 256, 0, stream>>>(pred_bboxes, gt_bboxes, metrics, ingts, bestm, A, M);

  k3_topk<<<B * M, 256, 0, stream>>>(metrics, ingts, intopk, A, M);

  k4_assign<<<g2, 256, 0, stream>>>(pred_bboxes, gt_bboxes, gt_labels, bg_index,
                                    metrics, ingts, intopk, bestm, maskpos,
                                    maxmet, maxiou, out_labels, out_bboxes, A, M);

  k5_scores<<<g2, 256, 0, stream>>>(maskpos, metrics, maxmet, maxiou,
                                    out_labels, out_scores, A, C, M);
}
